// WindowAttention_7456063225896
// MI455X (gfx1250) — matrix operand order repaired and checked by reading, not compiled, not run on hardware
//
#include <hip/hip_runtime.h>
#include <hip/hip_bf16.h>
#include <stdint.h>

typedef __attribute__((ext_vector_type(16))) __bf16 v16bf;
typedef __attribute__((ext_vector_type(8)))  __bf16 v8bf;
typedef __attribute__((ext_vector_type(2)))  __bf16 v2bf;
typedef __attribute__((ext_vector_type(8)))  float  v8f;

#define C_DIM   128
#define N_QKV   384
#define NH      4
#define HD      32
#define T_VALID 49
#define TP      64
#define IMG     222
#define WS7     7
#define SCALE_Q 0.17677669529663687f   // 1/sqrt(32)
#define NEGBIG  -3.0e38f

// LDS layout (bytes). Total 96 KB dynamic.
#define XT_OFF 0        // x tile bf16 [64][128]; reused later as O [64][128]
#define Q_OFF  16384    // bf16 [4][64][32]
#define K_OFF  32768    // bf16 [4][64][32]
#define V_OFF  49152    // bf16 [4][32][64]  (transposed: [head][d][token])
#define P_OFF  65536    // bf16 [4][64][64]
#define LDS_BYTES 98304

__device__ __forceinline__ v16bf load_frag_a(const __bf16* p) {
  // A-fragment: two 8-element (16B) chunks, 16 elements apart in K
  v8bf lo = *(const v8bf*)p;
  v8bf hi = *(const v8bf*)(p + 16);
  return __builtin_shufflevector(lo, hi, 0,1,2,3,4,5,6,7,8,9,10,11,12,13,14,15);
}

__device__ __forceinline__ float hmax16(float v) {
  #pragma unroll
  for (int off = 1; off < 16; off <<= 1) v = fmaxf(v, __shfl_xor(v, off, 32));
  return v;
}
__device__ __forceinline__ float hsum16(float v) {
  #pragma unroll
  for (int off = 1; off < 16; off <<= 1) v += __shfl_xor(v, off, 32);
  return v;
}

__global__ void prep_weights(const float* __restrict__ qkv_w, const float* __restrict__ proj_w,
                             __bf16* __restrict__ qkv_wt, __bf16* __restrict__ proj_wt) {
  int i = blockIdx.x * blockDim.x + threadIdx.x;
  if (i < N_QKV * C_DIM) {                 // qkv_wt[n][k] = qkv_w[k][n], bf16
    int n = i >> 7, k = i & 127;
    qkv_wt[i] = (__bf16)qkv_w[k * N_QKV + n];
  } else if (i < (N_QKV + C_DIM) * C_DIM) { // proj_wt[n][k] = proj_w[k][n], bf16
    int j = i - N_QKV * C_DIM;
    int n = j >> 7, k = j & 127;
    proj_wt[j] = (__bf16)proj_w[k * C_DIM + n];
  }
}

__launch_bounds__(256)
__global__ void win_attn_kernel(const float* __restrict__ x,
                                const float* __restrict__ qkv_b,
                                const float* __restrict__ proj_b,
                                const __bf16* __restrict__ qkv_wt,
                                const __bf16* __restrict__ proj_wt,
                                float* __restrict__ out) {
  extern __shared__ char smem[];
  __bf16* xt   = (__bf16*)(smem + XT_OFF);  // also O after phase 3
  __bf16* Qs   = (__bf16*)(smem + Q_OFF);
  __bf16* Ks   = (__bf16*)(smem + K_OFF);
  __bf16* Vt   = (__bf16*)(smem + V_OFF);
  __bf16* Ps   = (__bf16*)(smem + P_OFF);

  const int tid   = threadIdx.x;
  const int wave  = tid >> 5;
  const int lane  = tid & 31;
  const int ln    = lane & 15;      // N column within tile / A row within tile
  const int lhalf = lane >> 4;

  const int win = blockIdx.x;
  const int b   = win >> 10;
  const int wh  = (win >> 5) & 31;
  const int ww  = win & 31;
  const int h0  = wh * WS7;
  const int w0  = ww * WS7;

  // ---------------- Phase 0: stage window tile into LDS (bf16) ----------------
  {
    uint32_t* z = (uint32_t*)(xt + T_VALID * C_DIM);     // zero pad rows 49..63
    for (int i = tid; i < (TP - T_VALID) * C_DIM / 2; i += 256) z[i] = 0u;
    for (int i = tid; i < T_VALID * (C_DIM / 4); i += 256) {
      int t  = i >> 5;            // token 0..48
      int c4 = i & 31;            // float4 index within row
      int hi = h0 + t / WS7, wi = w0 + t % WS7;
      float4 v = make_float4(0.f, 0.f, 0.f, 0.f);
      if (hi < IMG && wi < IMG)
        v = *(const float4*)(x + (((size_t)b * IMG + hi) * IMG + wi) * C_DIM + c4 * 4);
      __bf16* d = xt + t * C_DIM + c4 * 4;
      d[0] = (__bf16)v.x; d[1] = (__bf16)v.y; d[2] = (__bf16)v.z; d[3] = (__bf16)v.w;
    }
  }
  __syncthreads();

  // ---------------- Phase 1: QKV GEMM [64x128] @ [128x384] ----------------
  {
    const int mt = wave >> 1;                 // 0..3
    const int ntbase = (wave & 1) * 12;       // split 24 n-tiles across wave pairs
    v16bf a[4];
    #pragma unroll
    for (int kt = 0; kt < 4; ++kt)
      a[kt] = load_frag_a(xt + (mt * 16 + ln) * C_DIM + kt * 32 + lhalf * 8);

    for (int j = 0; j < 12; ++j) {
      const int nt = ntbase + j;
      v8f acc = {};
      #pragma unroll
      for (int kt = 0; kt < 4; ++kt) {
        v16bf bfrag = load_frag_a(qkv_wt + (nt * 16 + ln) * C_DIM + kt * 32 + lhalf * 8);
        acc = __builtin_amdgcn_wmma_f32_16x16x32_bf16(false, a[kt], false, bfrag,
                                                      (short)0, acc, false, false);
      }
      const int n    = nt * 16 + ln;          // 0..383
      const float bias = qkv_b[n];
      // which / head are wave-uniform (a 16-wide tile never crosses a head
      // boundary) -> force them scalar so the compiler emits s_cbranch, not
      // exec-masked divergence, inside the hot GEMM loop.
      const int which = __builtin_amdgcn_readfirstlane(nt >> 3);        // 0=Q 1=K 2=V
      const int head  = __builtin_amdgcn_readfirstlane((nt >> 1) & 3);
      const int d     = ((nt & 1) << 4) + ln;                           // 0..31
      if (which == 0) {
        #pragma unroll
        for (int r = 0; r < 8; ++r) {
          int m = mt * 16 + r + 8 * lhalf;
          Qs[head * 2048 + m * HD + d] = (__bf16)((acc[r] + bias) * SCALE_Q);
        }
      } else if (which == 1) {
        #pragma unroll
        for (int r = 0; r < 8; ++r) {
          int m = mt * 16 + r + 8 * lhalf;
          Ks[head * 2048 + m * HD + d] = (__bf16)(acc[r] + bias);
        }
      } else {
        #pragma unroll
        for (int r = 0; r < 8; r += 2) {      // paired store along token dim
          int m = mt * 16 + r + 8 * lhalf;
          v2bf p; p[0] = (__bf16)(acc[r] + bias); p[1] = (__bf16)(acc[r + 1] + bias);
          *(v2bf*)(Vt + head * 2048 + d * TP + m) = p;
        }
      }
    }
  }
  __syncthreads();

  // ---------------- Phase 2: S = Q K^T, mask, softmax -> P (bf16 LDS) ----------------
  {
    const int head  = wave >> 1;
    const int mhalf = wave & 1;               // rows mhalf*32 .. +31
    v8f s[2][4];
    #pragma unroll
    for (int mt2 = 0; mt2 < 2; ++mt2) {
      v16bf aq = load_frag_a(Qs + head * 2048 + (mhalf * 32 + mt2 * 16 + ln) * HD + lhalf * 8);
      #pragma unroll
      for (int nt = 0; nt < 4; ++nt) {
        v16bf bk = load_frag_a(Ks + head * 2048 + (nt * 16 + ln) * HD + lhalf * 8);
        v8f z = {};
        s[mt2][nt] = __builtin_amdgcn_wmma_f32_16x16x32_bf16(false, aq, false, bk,
                                                             (short)0, z, false, false);
      }
    }
    #pragma unroll
    for (int mt2 = 0; mt2 < 2; ++mt2) {
      #pragma unroll
      for (int r = 0; r < 8; ++r) {
        const int m = mhalf * 32 + mt2 * 16 + r + 8 * lhalf;
        float v[4];
        float mx = NEGBIG;
        #pragma unroll
        for (int nt = 0; nt < 4; ++nt) {
          float t = s[mt2][nt][r];
          if (nt * 16 + ln >= T_VALID) t = NEGBIG;   // mask pad columns
          v[nt] = t;
          mx = fmaxf(mx, t);
        }
        mx = hmax16(mx);
        float e[4], sum = 0.f;
        #pragma unroll
        for (int nt = 0; nt < 4; ++nt) { e[nt] = __expf(v[nt] - mx); sum += e[nt]; }
        sum = hsum16(sum);
        const float inv = 1.f / sum;
        #pragma unroll
        for (int nt = 0; nt < 4; ++nt)
          Ps[head * 4096 + m * TP + nt * 16 + ln] = (__bf16)(e[nt] * inv);
      }
    }
  }
  __syncthreads();

  // ---------------- Phase 3: O = P V  (heads concatenated into xt region) ----------------
  {
    const int head  = wave >> 1;
    const int mhalf = wave & 1;
    v8f o[2][2] = {};
    #pragma unroll
    for (int mt2 = 0; mt2 < 2; ++mt2) {
      #pragma unroll
      for (int kt = 0; kt < 2; ++kt) {
        v16bf ap = load_frag_a(Ps + head * 4096 + (mhalf * 32 + mt2 * 16 + ln) * TP
                               + kt * 32 + lhalf * 8);
        #pragma unroll
        for (int nt = 0; nt < 2; ++nt) {
          v16bf bv = load_frag_a(Vt + head * 2048 + (nt * 16 + ln) * TP
                                     + kt * 32 + lhalf * 8);
          o[mt2][nt] = __builtin_amdgcn_wmma_f32_16x16x32_bf16(false, ap, false, bv,
                                                               (short)0, o[mt2][nt],
                                                               false, false);
        }
      }
    }
    #pragma unroll
    for (int mt2 = 0; mt2 < 2; ++mt2)
      #pragma unroll
      for (int nt = 0; nt < 2; ++nt)
        #pragma unroll
        for (int r = 0; r < 8; ++r) {
          int m = mhalf * 32 + mt2 * 16 + r + 8 * lhalf;
          xt[m * C_DIM + head * HD + nt * 16 + ln] = (__bf16)o[mt2][nt][r];
        }
  }
  __syncthreads();

  // ---------------- Phase 4: projection + bias + window-reverse store ----------------
  {
    const int mt = wave >> 1;
    const int ntbase = (wave & 1) * 4;
    v16bf a[4];
    #pragma unroll
    for (int kt = 0; kt < 4; ++kt)
      a[kt] = load_frag_a(xt + (mt * 16 + ln) * C_DIM + kt * 32 + lhalf * 8);

    #pragma unroll
    for (int j = 0; j < 4; ++j) {
      const int nt = ntbase + j;
      v8f acc = {};
      #pragma unroll
      for (int kt = 0; kt < 4; ++kt) {
        v16bf bfrag = load_frag_a(proj_wt + (nt * 16 + ln) * C_DIM + kt * 32 + lhalf * 8);
        acc = __builtin_amdgcn_wmma_f32_16x16x32_bf16(false, a[kt], false, bfrag,
                                                      (short)0, acc, false, false);
      }
      const int n = nt * 16 + ln;
      const float bias = proj_b[n];
      #pragma unroll
      for (int r = 0; r < 8; ++r) {
        int m = mt * 16 + r + 8 * lhalf;      // token index
        if (m < T_VALID) {
          int hi = h0 + m / WS7, wi = w0 + m % WS7;
          if (hi < IMG && wi < IMG)
            out[(((size_t)b * IMG + hi) * IMG + wi) * C_DIM + n] = acc[r] + bias;
        }
      }
    }
  }
}

extern "C" void kernel_launch(void* const* d_in, const int* in_sizes, int n_in,
                              void* d_out, int out_size, void* d_ws, size_t ws_size,
                              hipStream_t stream) {
  (void)in_sizes; (void)n_in; (void)out_size; (void)ws_size;
  const float* x      = (const float*)d_in[0];
  const float* qkv_w  = (const float*)d_in[1];
  const float* qkv_b  = (const float*)d_in[2];
  const float* proj_w = (const float*)d_in[3];
  const float* proj_b = (const float*)d_in[4];
  float* out = (float*)d_out;

  __bf16* qkv_wt  = (__bf16*)d_ws;                                  // 384*128 bf16
  __bf16* proj_wt = (__bf16*)((char*)d_ws + N_QKV * C_DIM * 2);     // 128*128 bf16

  prep_weights<<<(N_QKV * C_DIM + C_DIM * C_DIM + 255) / 256, 256, 0, stream>>>(
      qkv_w, proj_w, qkv_wt, proj_wt);

  win_attn_kernel<<<8192, 256, LDS_BYTES, stream>>>(
      x, qkv_b, proj_b, qkv_wt, proj_wt, out);
}
